// ConvInfoGathererLayer_63831803953665
// MI455X (gfx1250) — compile-verified, matrix operand order checked by reading, not run on hardware
//
#include <hip/hip_runtime.h>
#include <hip/hip_bf16.h>
#include <math.h>

typedef __attribute__((ext_vector_type(16))) _Float16 v16h;
typedef __attribute__((ext_vector_type(8)))  float    v8f;

#define B_  8
#define S_  32
#define E_  16
#define H_  2
#define F_  5
#define V_  256
#define LF_ 4
#define CF_ 128

__device__ __forceinline__ float frelu(float x) { return fmaxf(x, 0.f); }

// tanh(x) = (e-1)*rcp(e+1), e = 2^(2x*log2e). Native v_exp_f32 + v_rcp_f32.
__device__ __forceinline__ float fast_tanh(float x) {
  x = fminf(15.f, fmaxf(-15.f, x));
  float e = __builtin_amdgcn_exp2f(x * 2.8853900817779268f); // 2*log2(e)
  return (e - 1.f) * __builtin_amdgcn_rcpf(e + 1.f);
}

// One wave32 per (h, b, s) instance; 8 waves per block (same h,b; s varies);
// grid = 64 blocks.
__global__ __launch_bounds__(256)
void ConvInfoGathererLayer_kernel(
    const float* __restrict__ infovecs, const float* __restrict__ sequence,
    const float* __restrict__ Wc0, const float* __restrict__ bc0,
    const float* __restrict__ Wk0, const float* __restrict__ bk0,
    const float* __restrict__ Wc1, const float* __restrict__ bc1,
    const float* __restrict__ Wk1, const float* __restrict__ bk1,
    const float* __restrict__ Wc2, const float* __restrict__ bc2,
    const float* __restrict__ Wk2, const float* __restrict__ bk2,
    const float* __restrict__ Wdc, const float* __restrict__ bdc,
    const float* __restrict__ Wdk, const float* __restrict__ bdk,
    float* __restrict__ out)
{
  __shared__ float ybuf[8][2][512];   // per-wave ping-pong for y_j
  __shared__ float seqbuf[512];       // sequence[b] staged once per block

  const int lane = threadIdx.x & 31;
  const int wib  = threadIdx.x >> 5;
  const int gw   = blockIdx.x * 8 + wib;       // 0..511
  const int h = gw >> 8;                       // H=2
  const int b = (gw >> 5) & 7;                 // B=8
  const int s = gw & 31;                       // S=32

  // ---- stage sequence[b,:,:] (512 floats, shared by all waves in block) ----
  {
    const int t = threadIdx.x;
    seqbuf[t]       = sequence[b * 512 + t];
    seqbuf[t + 256] = sequence[b * 512 + t + 256];
  }
  __syncthreads();

  // ---- hypernet choke scalars (EC == 1): redundant per lane, 4x16 MACs ----
  float iv[E_];
  #pragma unroll
  for (int e = 0; e < E_; ++e) iv[e] = infovecs[(b * S_ + s) * E_ + e];

  const float* Wcs[3] = {Wc0, Wc1, Wc2};
  const float* bcs[3] = {bc0, bc1, bc2};
  float ch[3], chd;
  #pragma unroll
  for (int j = 0; j < 3; ++j) {
    float a = 0.f;
    #pragma unroll
    for (int e = 0; e < E_; ++e) a += iv[e] * Wcs[j][h * E_ + e];
    ch[j] = frelu(a + bcs[j][h]);
  }
  {
    float a = 0.f;
    #pragma unroll
    for (int e = 0; e < E_; ++e) a += iv[e] * Wdc[h * E_ + e];
    chd = frelu(a + bdc[h]);
  }

  // WMMA per-lane geometry (cdna5 16-bit fragment layout, wave32)
  const int mrow  = lane & 15;                 // A: M row / B: N col
  const int kbase = (lane < 16) ? 0 : 8;

  const float* Wks[3] = {Wk0, Wk1, Wk2};
  const float* bks[3] = {bk0, bk1, bk2};
  const int cins[3]  = {16, 32, 64};
  const int couts[3] = {32, 64, 128};
  const int lins[3]  = {32, 16, 8};
  const int louts[3] = {16, 8, 4};

  const float* prev = seqbuf;                  // layer-0 input lives in LDS too
  #pragma unroll
  for (int j = 0; j < 3; ++j) {
    const int cin  = cins[j],  cout = couts[j];
    const int lin  = lins[j],  lout = louts[j];
    const int Kv   = F_ * cin;                 // 80, 160, 320
    const int ktn  = (Kv + 31) >> 5;           // 3, 5, 10
    const int ntn  = cout >> 4;                // 2, 4, 8
    const float* Wk = Wks[j] + h * (Kv * cout);
    const float* bk = bks[j] + h * (Kv * cout);
    float* cur = &ybuf[wib][j & 1][0];
    const float chj = ch[j];

    // ---- build all A fragments once (register-resident; reused per N-tile).
    // Unconditional clamped LDS loads + value-select: no exec branching.
    v16h afr[10];
    #pragma unroll
    for (int kt = 0; kt < ktn; ++kt) {
      #pragma unroll
      for (int v = 0; v < 16; ++v) {
        const int k   = kt * 32 + ((v >> 3) << 4) + kbase + (v & 7);
        const int f   = k / cin;               // cin = 2^p -> shifts
        const int ci  = k - f * cin;
        const int row = 2 * mrow + f - 1;
        const bool ok = (k < Kv) && (row >= 0) && (row < lin);
        const int idx = ok ? (row * cin + ci) : 0;
        const float val = prev[idx];
        afr[kt][v] = (_Float16)(ok ? val : 0.f);
      }
    }

    for (int nt = 0; nt < ntn; ++nt) {
      const int n = nt * 16 + mrow;
      v8f acc = {};
      #pragma unroll
      for (int kt = 0; kt < ktn; ++kt) {
        // ---- B fragment: hyper-generated kernel, tanh on the fly ----
        v16h bf;
        #pragma unroll
        for (int v = 0; v < 16; ++v) {
          const int k  = kt * 32 + ((v >> 3) << 4) + kbase + (v & 7);
          const int kc = (k < Kv) ? k : (Kv - 1);        // clamp, load always
          const int idx = kc * cout + n;
          float t = fast_tanh(chj * Wk[idx] + bk[idx]);
          bf[v] = (_Float16)((k < Kv) ? t : 0.f);
        }
        acc = __builtin_amdgcn_wmma_f32_16x16x32_f16(
                  false, afr[kt], false, bf, (short)0, acc, false, false);
      }
      // ---- relu + store valid D rows to LDS ----
      #pragma unroll
      for (int r = 0; r < 8; ++r) {
        const int m = r + ((lane >> 4) << 3);
        if (m < lout) cur[m * cout + n] = frelu(acc[r]);
      }
    }
    __syncthreads();
    prev = cur;
  }

  // ---- per-instance dense head: out[v] = relu(sum_k yf[k]*tanh(chd*W+b)) ----
  const float* yf = &ybuf[wib][0][0];          // y2 flat [LF*CF] = [512]
  const float* Wd = Wdk + h * (LF_ * CF_ * V_);
  const float* bd = bdk + h * (LF_ * CF_ * V_);
  float acc[8] = {0.f, 0.f, 0.f, 0.f, 0.f, 0.f, 0.f, 0.f};
  for (int k = 0; k < LF_ * CF_; ++k) {
    const float yv = yf[k];                    // LDS broadcast (uniform addr)
    #pragma unroll
    for (int i = 0; i < 8; ++i) {
      const int idx = k * V_ + lane + 32 * i;  // coalesced across lanes
      acc[i] += yv * fast_tanh(chd * Wd[idx] + bd[idx]);
    }
  }
  #pragma unroll
  for (int i = 0; i < 8; ++i)
    out[((b * S_ + s) * H_ + h) * V_ + lane + 32 * i] = frelu(acc[i]);
}

extern "C" void kernel_launch(void* const* d_in, const int* in_sizes, int n_in,
                              void* d_out, int out_size, void* d_ws, size_t ws_size,
                              hipStream_t stream) {
  (void)in_sizes; (void)n_in; (void)out_size; (void)d_ws; (void)ws_size;
  const float* infovecs = (const float*)d_in[0];
  const float* sequence = (const float*)d_in[1];
  const float* Wc0 = (const float*)d_in[2];
  const float* bc0 = (const float*)d_in[3];
  const float* Wk0 = (const float*)d_in[4];
  const float* bk0 = (const float*)d_in[5];
  const float* Wc1 = (const float*)d_in[6];
  const float* bc1 = (const float*)d_in[7];
  const float* Wk1 = (const float*)d_in[8];
  const float* bk1 = (const float*)d_in[9];
  const float* Wc2 = (const float*)d_in[10];
  const float* bc2 = (const float*)d_in[11];
  const float* Wk2 = (const float*)d_in[12];
  const float* bk2 = (const float*)d_in[13];
  const float* Wdc = (const float*)d_in[14];
  const float* bdc = (const float*)d_in[15];
  const float* Wdk = (const float*)d_in[16];
  const float* bdk = (const float*)d_in[17];
  float* out = (float*)d_out;

  // 512 instances (H*B*S), one wave32 each, 8 waves per block -> 64 blocks
  ConvInfoGathererLayer_kernel<<<64, 256, 0, stream>>>(
      infovecs, sequence, Wc0, bc0, Wk0, bk0, Wc1, bc1, Wk1, bk1,
      Wc2, bc2, Wk2, bk2, Wdc, bdc, Wdk, bdk, out);
}
